// TransformerAttention_83494164234866
// MI455X (gfx1250) — matrix operand order repaired and checked by reading, not compiled, not run on hardware
//
#include <hip/hip_runtime.h>

// Problem constants (fixed by the reference)
#define BQ     2
#define NQ     2048
#define DIMC   1024
#define NHEADS 8
#define DHC    64
#define NMEM   4
#define JPAD   2080          // 4 mem + 2048 keys, padded to multiple of 32
#define NEGF   (-1e30f)

typedef __attribute__((ext_vector_type(8)))  __bf16 v8bf;
typedef __attribute__((ext_vector_type(16))) __bf16 v16bf;
typedef __attribute__((ext_vector_type(8)))  float  v8f;
typedef __attribute__((ext_vector_type(4)))  int    v4i;

// ---- async global->LDS (CDNA5 ASYNCcnt path), guarded for toolchain support
#if defined(__has_builtin)
#  if __has_builtin(__builtin_amdgcn_global_load_async_to_lds_b128) && \
      __has_builtin(__builtin_amdgcn_s_wait_asynccnt)
#    define HAVE_ASYNC 1
#  endif
#endif
#ifndef HAVE_ASYNC
#  define HAVE_ASYNC 0
#endif

static __device__ __forceinline__ void copy16(const __bf16* __restrict__ g,
                                              __bf16* __restrict__ l) {
#if HAVE_ASYNC
  // param type (from hipcc diagnostic): v4i* for both src and LDS dst
  __builtin_amdgcn_global_load_async_to_lds_b128((v4i*)(void*)g, (v4i*)(void*)l, 0, 0);
#else
  *(v8bf*)l = *(const v8bf*)g;
#endif
}
static __device__ __forceinline__ void wait_async() {
#if HAVE_ASYNC
  __builtin_amdgcn_s_wait_asynccnt(0);
#endif
}

static __device__ __forceinline__ v16bf cat16(v8bf lo, v8bf hi) {
  return __builtin_shufflevector(lo, hi, 0,1,2,3,4,5,6,7,8,9,10,11,12,13,14,15);
}
static __device__ __forceinline__ v8f wmma_bf16(v16bf a, v16bf b, v8f c) {
  // D = A(16x32 bf16) * B(32x16 bf16) + C(16x16 f32)
  return __builtin_amdgcn_wmma_f32_16x16x32_bf16(false, a, false, b, (short)0, c, false, false);
}

// ---- DPP16 cross-lane reductions within each 16-lane row (wave32 = 2 rows) --
template <int CTRL>
static __device__ __forceinline__ float dppmov(float x) {
  return __builtin_bit_cast(float, __builtin_amdgcn_update_dpp(
      0, __builtin_bit_cast(int, x), CTRL, 0xF, 0xF, true));
}
static __device__ __forceinline__ float rowmax16(float x) {
  x = fmaxf(x, dppmov<0xB1>(x));    // quad_perm xor1
  x = fmaxf(x, dppmov<0x4E>(x));    // quad_perm xor2
  x = fmaxf(x, dppmov<0x141>(x));   // row_half_mirror (xor7)
  x = fmaxf(x, dppmov<0x140>(x));   // row_mirror (xor15)
  return x;
}
static __device__ __forceinline__ float rowsum16(float x) {
  x += dppmov<0xB1>(x);
  x += dppmov<0x4E>(x);
  x += dppmov<0x141>(x);
  x += dppmov<0x140>(x);
  return x;
}

// ---------------------------------------------------------------------------
// Kernel 1: row LayerNorm + bf16 conversion of x and xn.  grid = B*N rows.
// ---------------------------------------------------------------------------
__global__ void __launch_bounds__(256) prep_ln(
    const float* __restrict__ x, const float* __restrict__ gamma,
    __bf16* __restrict__ xbf, __bf16* __restrict__ xnbf) {
  const int row = blockIdx.x;                 // 0 .. B*N-1
  const int t   = threadIdx.x;                // 256 threads
  const float* xr = x + (size_t)row * DIMC;
  float v[4], s = 0.f, ss = 0.f;
#pragma unroll
  for (int i = 0; i < 4; ++i) {
    v[i] = xr[t + i * 256];
    s  += v[i];
    ss += v[i] * v[i];
  }
#pragma unroll
  for (int off = 16; off > 0; off >>= 1) {
    s  += __shfl_xor(s,  off, 32);
    ss += __shfl_xor(ss, off, 32);
  }
  __shared__ float sred[8], ssred[8];
  if ((t & 31) == 0) { sred[t >> 5] = s; ssred[t >> 5] = ss; }
  __syncthreads();
  float st = 0.f, sst = 0.f;
#pragma unroll
  for (int w = 0; w < 8; ++w) { st += sred[w]; sst += ssred[w]; }
  const float mu  = st * (1.f / DIMC);
  const float var = sst * (1.f / DIMC) - mu * mu;
  const float rs  = rsqrtf(var + 1e-5f);
#pragma unroll
  for (int i = 0; i < 4; ++i) {
    const int c = t + i * 256;
    xbf [(size_t)row * DIMC + c] = (__bf16)v[i];
    xnbf[(size_t)row * DIMC + c] = (__bf16)((v[i] - mu) * rs * gamma[c]);
  }
}

// ---------------------------------------------------------------------------
// Kernel 2: f32 -> bf16 weight conversion WITH transpose:
//   in[r][c] (rows x cols, row-major)  ->  outT[c][r]
// B-tiles of the GEMM then become contiguous [col][k] rows (async-friendly).
// ---------------------------------------------------------------------------
__global__ void __launch_bounds__(256) convT_bf16(
    const float* __restrict__ in, __bf16* __restrict__ outT, int rows, int cols) {
  int i = blockIdx.x * 256 + threadIdx.x;
  if (i < rows * cols) {
    const int r = i / cols, c = i % cols;
    outT[(size_t)c * rows + r] = (__bf16)in[i];
  }
}

// ---------------------------------------------------------------------------
// Kernel 3: mem-KV / padding / additive-mask fixup over (b, j) in [B][JPAD]
// ---------------------------------------------------------------------------
__global__ void __launch_bounds__(256) fixup(
    const float* __restrict__ mem_kv, const int* __restrict__ mask,
    __bf16* __restrict__ kbuf, __bf16* __restrict__ vt, float* __restrict__ mbuf) {
  int idx = blockIdx.x * 256 + threadIdx.x;
  if (idx >= BQ * JPAD) return;
  const int b = idx / JPAD, j = idx % JPAD;
  float mv;
  if (j < NMEM)               mv = 0.f;
  else if (j < NMEM + NQ)     mv = (mask[b * NQ + (j - NMEM)] != 0) ? 0.f : NEGF;
  else                        mv = NEGF;
  mbuf[idx] = mv;
  if (j < NMEM) {
#pragma unroll
    for (int d = 0; d < DHC; ++d) {
      kbuf[((size_t)b * JPAD + j) * DHC + d] = (__bf16)mem_kv[j * DHC + d];              // mem_kv[0]
      vt  [((size_t)b * DHC + d) * JPAD + j] = (__bf16)mem_kv[NMEM * DHC + j * DHC + d]; // mem_kv[1]
    }
  } else if (j >= NMEM + NQ) {
#pragma unroll
    for (int d = 0; d < DHC; ++d) {
      kbuf[((size_t)b * JPAD + j) * DHC + d] = (__bf16)0.f;
      vt  [((size_t)b * DHC + d) * JPAD + j] = (__bf16)0.f;
    }
  }
}

// ---------------------------------------------------------------------------
// Kernel 4: double-buffered bf16 WMMA GEMM: C(MxNc) = alpha * A(MxK) @ Bt^T
//   A: MxK row-major.  Bt: Nc x K (pre-transposed weights) -> both tiles are
//   contiguous b128 rows, streamed global->LDS via async copies (ASYNCcnt).
//   block = 256 threads = 8 waves; tile 128x128, K-step 32.
// MODE 0: fp32 row-major     MODE 1: q layout (B,H,N,DH) bf16
// MODE 2: cols<64 -> kbuf[b][4+n][c], cols>=64 -> vT[b][c-64][4+n]
// ---------------------------------------------------------------------------
template <int MODE>
__global__ void __launch_bounds__(256) gemm_kernel(
    const __bf16* __restrict__ A, const __bf16* __restrict__ Bt,
    void* __restrict__ Cout, void* __restrict__ Cout2,
    int M, int Ncols, int K, float alpha) {
  __shared__ __align__(16) __bf16 As[2][128 * 48];   // [row][k], stride 48
  __shared__ __align__(16) __bf16 Bs[2][128 * 48];   // [col][k], stride 48
  const int tid  = threadIdx.x;
  const int lane = tid & 31;
  const int col  = lane & 15;
  const int hi   = lane >> 4;
  const int wave = tid >> 5;
  const int wm   = (wave & 3) * 32;
  const int wn   = (wave >> 2) * 64;
  const int bm   = blockIdx.x * 128;
  const int bn   = blockIdx.y * 128;

  v8f acc[2][4];
#pragma unroll
  for (int i = 0; i < 2; ++i)
#pragma unroll
    for (int j = 0; j < 4; ++j) acc[i][j] = v8f{0.f,0.f,0.f,0.f,0.f,0.f,0.f,0.f};

  // each thread copies 2 A-units and 2 B-units of 8 bf16 (b128) per tile
  const int r0 = tid >> 2,           cg0 = (tid & 3) * 8;
  const int r1 = (tid + 256) >> 2,   cg1 = (tid & 3) * 8;   // (tid+256)&3 == tid&3
  auto load_tiles = [&](int k0, int pbuf) {
    __bf16* Ab = As[pbuf];
    __bf16* Bb = Bs[pbuf];
    copy16(A  + (size_t)(bm + r0) * K + k0 + cg0, Ab + r0 * 48 + cg0);
    copy16(A  + (size_t)(bm + r1) * K + k0 + cg1, Ab + r1 * 48 + cg1);
    copy16(Bt + (size_t)(bn + r0) * K + k0 + cg0, Bb + r0 * 48 + cg0);
    copy16(Bt + (size_t)(bn + r1) * K + k0 + cg1, Bb + r1 * 48 + cg1);
  };

  load_tiles(0, 0);
  int buf = 0;
  for (int k0 = 0; k0 < K; k0 += 32) {
    wait_async();          // this wave's async copies into As/Bs[buf] done
    __syncthreads();       // everyone's copies done; prev reads of buf^1 done
    if (k0 + 32 < K) load_tiles(k0 + 32, buf ^ 1);

    const __bf16* Ab = As[buf];
    const __bf16* Bb = Bs[buf];
    v16bf af[2], bf[4];
#pragma unroll
    for (int i = 0; i < 2; ++i) {
      const __bf16* ap = Ab + (wm + i * 16 + col) * 48 + hi * 16; // K = hi*16 + 0..15
      af[i] = cat16(*(const v8bf*)ap, *(const v8bf*)(ap + 8));
    }
#pragma unroll
    for (int j = 0; j < 4; ++j) {
      const __bf16* bp = Bb + (wn + j * 16 + col) * 48 + hi * 16; // K = hi*16 + 0..15
      bf[j] = cat16(*(const v8bf*)bp, *(const v8bf*)(bp + 8));
    }
#pragma unroll
    for (int i = 0; i < 2; ++i)
#pragma unroll
      for (int j = 0; j < 4; ++j)
        acc[i][j] = wmma_bf16(af[i], bf[j], acc[i][j]);
    buf ^= 1;
  }

#pragma unroll
  for (int i = 0; i < 2; ++i)
#pragma unroll
    for (int j = 0; j < 4; ++j)
#pragma unroll
      for (int v = 0; v < 8; ++v) {
        const int r = bm + wm + i * 16 + v + hi * 8;
        const int c = bn + wn + j * 16 + col;
        const float val = acc[i][j][v] * alpha;
        if (MODE == 0) {
          ((float*)Cout)[(size_t)r * Ncols + c] = val;
        } else if (MODE == 1) {              // q: (B,H,N,DH) bf16
          const int bb = r >> 11, n = r & (NQ - 1);
          const int hh = c >> 6,  dh = c & 63;
          ((__bf16*)Cout)[(((size_t)(bb * NHEADS + hh)) * NQ + n) * DHC + dh] = (__bf16)val;
        } else {                             // kv split
          const int bb = r >> 11, n = r & (NQ - 1);
          if (c < DHC)
            ((__bf16*)Cout)[((size_t)bb * JPAD + NMEM + n) * DHC + c] = (__bf16)val;
          else
            ((__bf16*)Cout2)[((size_t)bb * DHC + (c - DHC)) * JPAD + NMEM + n] = (__bf16)val;
        }
      }
}

// ---------------------------------------------------------------------------
// Kernel 5: flash attention.  block = 128 threads = 4 waves; wave = 16 q rows.
// grid = (N/64, B*H).  Keys streamed in 32-wide chunks up to causal bound.
// Softmax row stats via DPP16; bias loads branchless + non-temporal.
// ---------------------------------------------------------------------------
__global__ void __launch_bounds__(128) attn_kernel(
    const __bf16* __restrict__ qg, const __bf16* __restrict__ kbuf,
    const __bf16* __restrict__ vT, const float* __restrict__ bias,
    const float* __restrict__ mbuf, __bf16* __restrict__ aout) {
  const int lane = threadIdx.x & 31;
  const int col  = lane & 15;
  const int hi   = lane >> 4;
  const int wave = threadIdx.x >> 5;          // 0..3
  const int bh   = blockIdx.y;
  const int b    = bh / NHEADS, h = bh % NHEADS;
  const int qm   = blockIdx.x * 64 + wave * 16;

  __shared__ __align__(16) __bf16 plds[4][16 * 40];   // per-wave P bounce (stride 40)
  __bf16* pl = plds[wave];

  // q fragments (A layout): lane row = qm+col, K halves by hi
  const __bf16* qrow = qg + (((size_t)(b * NHEADS + h)) * NQ + (qm + col)) * DHC;
  const v16bf qa0 = cat16(*(const v8bf*)(qrow +       hi * 16), *(const v8bf*)(qrow + 8 + hi * 16));
  const v16bf qa1 = cat16(*(const v8bf*)(qrow + 32 +  hi * 16), *(const v8bf*)(qrow + 40 + hi * 16));

  const v8f z8 = v8f{0.f,0.f,0.f,0.f,0.f,0.f,0.f,0.f};
  v8f o0 = z8, o1 = z8, o2 = z8, o3 = z8;
  float mrun[8], lrun[8];
#pragma unroll
  for (int v = 0; v < 8; ++v) { mrun[v] = NEGF; lrun[v] = 0.f; }

  const int kend = ((qm + 15 + NMEM) >> 5) << 5;      // last chunk base (causal)
  for (int kb = 0; kb <= kend; kb += 32) {
    // K^T fragments (B layout): lane col = key, K(dh) = hi*16 + 0..15
    const __bf16* kr0 = kbuf + ((size_t)b * JPAD + kb +      col) * DHC + hi * 16;
    const __bf16* kr1 = kbuf + ((size_t)b * JPAD + kb + 16 + col) * DHC + hi * 16;
    const v16bf kf00 = cat16(*(const v8bf*)kr0,        *(const v8bf*)(kr0 + 8));
    const v16bf kf01 = cat16(*(const v8bf*)(kr0 + 32), *(const v8bf*)(kr0 + 40));
    const v16bf kf10 = cat16(*(const v8bf*)kr1,        *(const v8bf*)(kr1 + 8));
    const v16bf kf11 = cat16(*(const v8bf*)(kr1 + 32), *(const v8bf*)(kr1 + 40));

    v8f s0 = wmma_bf16(qa0, kf00, z8); s0 = wmma_bf16(qa1, kf01, s0);
    v8f s1 = wmma_bf16(qa0, kf10, z8); s1 = wmma_bf16(qa1, kf11, s1);

    // bias + key-mask + causal + online softmax, fused per row (C layout:
    // lane = key col -> coalesced bias reads along the key axis)
    const int j0 = kb + col, j1 = kb + 16 + col;
    const float mv0 = mbuf[b * JPAD + j0], mv1 = mbuf[b * JPAD + j1];
    // branchless bias addressing: clamp into range, select after load
    const bool ok0 = (unsigned)(j0 - NMEM) < (unsigned)NQ;
    const bool ok1 = (unsigned)(j1 - NMEM) < (unsigned)NQ;
    const int jc0 = min(max(j0 - NMEM, 0), NQ - 1);
    const int jc1 = min(max(j1 - NMEM, 0), NQ - 1);
#pragma unroll
    for (int v = 0; v < 8; ++v) {
      const int qi = qm + v + hi * 8;
      const float* bp = bias + ((size_t)h * NQ + qi) * NQ;
      float b0 = __builtin_nontemporal_load(bp + jc0);
      float b1 = __builtin_nontemporal_load(bp + jc1);
      b0 = ok0 ? b0 : 0.f;
      b1 = ok1 ? b1 : 0.f;
      const float t0 = s0[v] + b0 + mv0 + ((j0 <= qi + NMEM) ? 0.f : NEGF);
      const float t1 = s1[v] + b1 + mv1 + ((j1 <= qi + NMEM) ? 0.f : NEGF);
      const float mx   = rowmax16(fmaxf(t0, t1));
      const float mnew = fmaxf(mrun[v], mx);
      const float sc = __expf(mrun[v] - mnew);
      const float e0 = __expf(t0 - mnew);
      const float e1 = __expf(t1 - mnew);
      const float rs = rowsum16(e0 + e1);
      lrun[v] = lrun[v] * sc + rs;
      mrun[v] = mnew;
      o0[v] *= sc; o1[v] *= sc; o2[v] *= sc; o3[v] *= sc;
      const int row = v + hi * 8;                      // re-layout P via LDS
      pl[row * 40 +      col] = (__bf16)e0;
      pl[row * 40 + 16 + col] = (__bf16)e1;
    }

    // P as A-fragment (K = 32 keys of this chunk)
    const __bf16* pr = pl + col * 40;
    const v16bf pa = cat16(*(const v8bf*)(pr + hi * 16), *(const v8bf*)(pr + 8 + hi * 16));

    // V fragments from vT (B layout): lane col = dh, 16 contiguous keys
#pragma unroll
    for (int d = 0; d < 4; ++d) {
      const __bf16* vr = vT + ((size_t)b * DHC + d * 16 + col) * JPAD + kb + hi * 16;
      const v16bf vf = cat16(*(const v8bf*)vr, *(const v8bf*)(vr + 8));
      if      (d == 0) o0 = wmma_bf16(pa, vf, o0);
      else if (d == 1) o1 = wmma_bf16(pa, vf, o1);
      else if (d == 2) o2 = wmma_bf16(pa, vf, o2);
      else             o3 = wmma_bf16(pa, vf, o3);
    }
  }

  // normalize and store attn-out as bf16 in (b*N + i, h*64 + dh) layout
#pragma unroll
  for (int v = 0; v < 8; ++v) {
    const float inv = 1.f / lrun[v];
    const int qi = qm + v + hi * 8;
    const size_t rowb = ((size_t)b * NQ + qi) * (NHEADS * DHC) + h * DHC;
    aout[rowb +      col] = (__bf16)(o0[v] * inv);
    aout[rowb + 16 + col] = (__bf16)(o1[v] * inv);
    aout[rowb + 32 + col] = (__bf16)(o2[v] * inv);
    aout[rowb + 48 + col] = (__bf16)(o3[v] * inv);
  }
}

// ---------------------------------------------------------------------------
extern "C" void kernel_launch(void* const* d_in, const int* in_sizes, int n_in,
                              void* d_out, int out_size, void* d_ws, size_t ws_size,
                              hipStream_t stream) {
  const float* x        = (const float*)d_in[0];
  const float* attn_bias= (const float*)d_in[1];
  const float* gamma    = (const float*)d_in[2];
  const float* Wq       = (const float*)d_in[3];
  const float* Wkv      = (const float*)d_in[4];
  const float* mem_kv   = (const float*)d_in[5];
  const float* Wout     = (const float*)d_in[6];
  const int*   mask     = (const int*)d_in[7];
  float* out = (float*)d_out;

  char* ws = (char*)d_ws;
  size_t off = 0;
  auto carve = [&](size_t bytes) { void* p = ws + off; off += (bytes + 255) & ~(size_t)255; return p; };
  __bf16* xbf    = (__bf16*)carve((size_t)BQ * NQ * DIMC * 2);
  __bf16* xnbf   = (__bf16*)carve((size_t)BQ * NQ * DIMC * 2);
  __bf16* WqT    = (__bf16*)carve((size_t)DIMC * NHEADS * DHC * 2);   // [512][1024]
  __bf16* WkvT   = (__bf16*)carve((size_t)DIMC * 2 * DHC * 2);        // [128][1024]
  __bf16* WoutT  = (__bf16*)carve((size_t)NHEADS * DHC * DIMC * 2);   // [1024][512]
  __bf16* qbf    = (__bf16*)carve((size_t)BQ * NHEADS * NQ * DHC * 2);
  __bf16* kbuf   = (__bf16*)carve((size_t)BQ * JPAD * DHC * 2);
  __bf16* vtb    = (__bf16*)carve((size_t)BQ * DHC * JPAD * 2);
  float*  mbuf   = (float*)carve((size_t)BQ * JPAD * 4);
  __bf16* aoutb  = (__bf16*)carve((size_t)BQ * NQ * NHEADS * DHC * 2);

  // 1) LayerNorm + bf16
  prep_ln<<<BQ * NQ, 256, 0, stream>>>(x, gamma, xbf, xnbf);
  // 2) weights -> bf16, transposed (Bt layouts for the GEMMs)
  convT_bf16<<<(DIMC * NHEADS * DHC + 255) / 256, 256, 0, stream>>>(Wq,  WqT,  DIMC, NHEADS * DHC);
  convT_bf16<<<(DIMC * 2 * DHC    + 255) / 256, 256, 0, stream>>>(Wkv, WkvT, DIMC, 2 * DHC);
  convT_bf16<<<(NHEADS * DHC * DIMC + 255) / 256, 256, 0, stream>>>(Wout, WoutT, NHEADS * DHC, DIMC);
  // 3) mem-kv / pad / mask fixup
  fixup<<<(BQ * JPAD + 255) / 256, 256, 0, stream>>>(mem_kv, mask, kbuf, vtb, mbuf);
  // 4) q = ln(x) @ Wq * DH^-0.5  -> (B,H,N,DH) bf16
  gemm_kernel<1><<<dim3((BQ * NQ) / 128, (NHEADS * DHC) / 128), 256, 0, stream>>>(
      xnbf, WqT, (void*)qbf, nullptr, BQ * NQ, NHEADS * DHC, DIMC, 0.125f);
  // 5) kv = x @ Wkv -> kbuf / vT (bf16)
  gemm_kernel<2><<<dim3((BQ * NQ) / 128, 1), 256, 0, stream>>>(
      xbf, WkvT, (void*)kbuf, (void*)vtb, BQ * NQ, 2 * DHC, DIMC, 1.0f);
  // 6) flash attention
  attn_kernel<<<dim3(NQ / 64, BQ * NHEADS), 128, 0, stream>>>(
      qbf, kbuf, vtb, attn_bias, mbuf, aoutb);
  // 7) out = attn_out @ Wout (fp32)
  gemm_kernel<0><<<dim3((BQ * NQ) / 128, DIMC / 128), 256, 0, stream>>>(
      aoutb, WoutT, (void*)out, nullptr, BQ * NQ, DIMC, NHEADS * DHC, 1.0f);
}
